// QuantLlamaAttentionFused_1726576853071
// MI455X (gfx1250) — compile-verified, matrix operand order checked by reading, not run on hardware
//
#include <hip/hip_runtime.h>

// ---------------------------------------------------------------------------
// Fused Llama attention for MI455X (gfx1250, wave32, WMMA).
// All GEMMs / einsums run on v_wmma_f32_16x16x32_bf16 with fp32 accumulation.
// GEMM stages tiles with GLOBAL_LOAD_ASYNC_TO_LDS_B128 (ASYNCcnt) and double
// buffering so global->LDS traffic overlaps the WMMA pipe.
// ---------------------------------------------------------------------------

typedef __bf16 bf16;
typedef __bf16 bf8v  __attribute__((ext_vector_type(8)));
typedef __bf16 v16bf __attribute__((ext_vector_type(16)));
typedef float  v8f   __attribute__((ext_vector_type(8)));

#define BATCH  2
#define SEQ    2048
#define NHEADS 32
#define HDIM   128
#define HID    4096

// ---------------------------------------------------------------------------
// Fragment loader matching the CDNA5 16-bit A-operand layout (CDNA5 ISA):
//   lane 0-15  : row = lane,    K = {0..7, 16..23}
//   lane 16-31 : row = lane-16, K = {8..15, 24..31}
// Works for both A (rows = M) and B (rows = N of W[N,K], i.e. B^T) operands.
// Two 16-byte loads per lane (ds_load_b128 / global_load_b128).
// ---------------------------------------------------------------------------
static __device__ __forceinline__ v16bf load_frag(const bf16* __restrict__ base, int ld) {
  const int lane = threadIdx.x & 31;
  const int r    = lane & 15;
  const int hi   = lane >> 4;
  const bf16* p  = base + (size_t)r * ld + hi * 8;
  bf8v a = *(const bf8v*)(p);
  bf8v b = *(const bf8v*)(p + 16);
  return __builtin_shufflevector(a, b, 0, 1, 2, 3, 4, 5, 6, 7,
                                       8, 9, 10, 11, 12, 13, 14, 15);
}

// Async 16-byte global->LDS copy (gfx1250 GLOBAL_LOAD_ASYNC_TO_LDS_B128).
// INST_OFFSET is added to BOTH the LDS and global addresses (ISA 08, §4.4),
// so consecutive chunks of one row can share the same address registers.
static __device__ __forceinline__ void async_b128(unsigned lds_addr, const void* gptr) {
  asm volatile("global_load_async_to_lds_b128 %0, %1, off"
               :: "v"(lds_addr), "v"((unsigned long long)(uintptr_t)gptr) : "memory");
}
static __device__ __forceinline__ void async_b128_off16(unsigned lds_addr, const void* gptr) {
  asm volatile("global_load_async_to_lds_b128 %0, %1, off offset:16"
               :: "v"(lds_addr), "v"((unsigned long long)(uintptr_t)gptr) : "memory");
}
static __device__ __forceinline__ unsigned lds_off(const void* p) {
  return (unsigned)(uintptr_t)p;   // low 32 bits of a flat shared pointer = LDS byte address
}

// ---------------------------------------------------------------------------
// fp32 -> bf16 elementwise convert (grid-stride)
// ---------------------------------------------------------------------------
__global__ void cvt_f32_to_bf16(const float* __restrict__ src,
                                bf16* __restrict__ dst, size_t n) {
  size_t i      = (size_t)blockIdx.x * blockDim.x + threadIdx.x;
  size_t stride = (size_t)gridDim.x * blockDim.x;
  for (; i < n; i += stride) dst[i] = (bf16)src[i];
}

// ---------------------------------------------------------------------------
// C[M,N] (fp32) = A[M,K] (bf16, row-major) @ W[N,K]^T (bf16, row-major)
// Block tile 128x64, BK=32, 8 waves (4x2), each wave 32x32 = 4 WMMA tiles.
// Double-buffered async global->LDS staging.
// ---------------------------------------------------------------------------
#define BM 128
#define BN 64
#define BK 32
#define LDP 40   // padded LDS row stride (halves); 80B rows keep 16B alignment

__global__ __launch_bounds__(256)
void gemm_bf16_nt(const bf16* __restrict__ A, const bf16* __restrict__ W,
                  float* __restrict__ C, int M, int N, int K) {
  __shared__ bf16 As[2][BM][LDP];
  __shared__ bf16 Bs[2][BN][LDP];

  const int t     = threadIdx.x;
  const int wid   = t >> 5;
  const int lane  = t & 31;
  const int waveM = wid >> 1;       // 0..3
  const int waveN = wid & 1;        // 0..1
  const int m0    = blockIdx.y * BM;
  const int n0    = blockIdx.x * BN;

  const int arow = t >> 1, acol = (t & 1) * 16;   // 128 rows x 32 cols, 16 halves/thread
  const int brow = t >> 2, bcol = (t & 3) * 8;    // 64 rows x 32 cols, 8 halves/thread

  const bf16* gA = A + (size_t)(m0 + arow) * K + acol;
  const bf16* gB = W + (size_t)(n0 + brow) * K + bcol;

  // per-stage: 3 async instructions per thread (A: 2x16B, B: 1x16B)
  auto issue_stage = [&](int kk, int buf) {
    const unsigned la = lds_off(&As[buf][arow][acol]);
    const unsigned lb = lds_off(&Bs[buf][brow][bcol]);
    const bf16* pa = gA + (size_t)kk * BK;
    const bf16* pb = gB + (size_t)kk * BK;
    async_b128(la, pa);
    async_b128_off16(la, pa);
    async_b128(lb, pb);
  };

  v8f acc[2][2] = {};
  const int NK = K / BK;
  issue_stage(0, 0);

  for (int kk = 0; kk < NK; ++kk) {
    const int buf = kk & 1;
    // this wave's stage-kk copies have landed in LDS
    asm volatile("s_wait_asynccnt 0x0" ::: "memory");
    // every wave's copies landed; every wave finished reading buf (its ds
    // reads were consumed by WMMAs behind s_wait_dscnt before this barrier)
    __syncthreads();
    if (kk + 1 < NK) {
      issue_stage(kk + 1, buf ^ 1);                         // overlap with compute
      __builtin_prefetch(gA + (size_t)(kk + 2) * BK, 0, 1); // warm L2 for stage kk+2
      __builtin_prefetch(gB + (size_t)(kk + 2) * BK, 0, 1);
    }

    v16bf af[2], bfg[2];
#pragma unroll
    for (int tm = 0; tm < 2; tm++) af[tm]  = load_frag(&As[buf][waveM * 32 + tm * 16][0], LDP);
#pragma unroll
    for (int tn = 0; tn < 2; tn++) bfg[tn] = load_frag(&Bs[buf][waveN * 32 + tn * 16][0], LDP);

#pragma unroll
    for (int tm = 0; tm < 2; tm++)
#pragma unroll
      for (int tn = 0; tn < 2; tn++)
        acc[tm][tn] = __builtin_amdgcn_wmma_f32_16x16x32_bf16(
            false, af[tm], false, bfg[tn], (short)0, acc[tm][tn], false, false);
  }

  // epilogue: C/D layout -> VGPR i: row = i + 8*(lane>>4), col = lane&15
  const int r = lane & 15, hi = lane >> 4;
#pragma unroll
  for (int tm = 0; tm < 2; tm++)
#pragma unroll
    for (int tn = 0; tn < 2; tn++)
#pragma unroll
      for (int i = 0; i < 8; i++) {
        int row = m0 + waveM * 32 + tm * 16 + i + 8 * hi;
        int col = n0 + waveN * 32 + tn * 16 + r;
        C[(size_t)row * N + col] = acc[tm][tn][i];
      }
}

// ---------------------------------------------------------------------------
// RoPE + reorder: qkv fp32 [B*S, 3*H] ->
//   Q,K bf16 [B,NH,S,D] (rope applied), V bf16 transposed [B,NH,D,S]
// One thread per (b,s,h,i) with i < D/2.
// ---------------------------------------------------------------------------
__global__ void rope_reorder(const float* __restrict__ qkv,
                             bf16* __restrict__ Q, bf16* __restrict__ Km,
                             bf16* __restrict__ Vt) {
  size_t idx = (size_t)blockIdx.x * blockDim.x + threadIdx.x;
  const size_t total = (size_t)BATCH * SEQ * NHEADS * (HDIM / 2);
  if (idx >= total) return;
  const int i = idx & 63;
  const int h = (idx >> 6) & 31;
  const int s = (idx >> 11) & 2047;
  const int b = (int)(idx >> 22);

  // inv_freq = theta^(-2i/D) = exp(-(2i/D) * ln(10000))
  float inv = __expf(-(2.0f * (float)i / (float)HDIM) * 9.2103403719761836f);
  float ang = (float)s * inv;
  float sn, c;
  __sincosf(ang, &sn, &c);

  const size_t row  = (size_t)b * SEQ + s;
  const float* base = qkv + row * (size_t)(3 * HID);
  const float q1 = base[0 * HID + h * HDIM + i];
  const float q2 = base[0 * HID + h * HDIM + i + 64];
  const float k1 = base[1 * HID + h * HDIM + i];
  const float k2 = base[1 * HID + h * HDIM + i + 64];
  const float v1 = base[2 * HID + h * HDIM + i];
  const float v2 = base[2 * HID + h * HDIM + i + 64];

  const size_t hb = (size_t)b * NHEADS + h;
  bf16* qo = Q  + hb * (size_t)SEQ * HDIM + (size_t)s * HDIM;
  bf16* ko = Km + hb * (size_t)SEQ * HDIM + (size_t)s * HDIM;
  qo[i]      = (bf16)(q1 * c - q2 * sn);
  qo[i + 64] = (bf16)(q1 * sn + q2 * c);
  ko[i]      = (bf16)(k1 * c - k2 * sn);
  ko[i + 64] = (bf16)(k1 * sn + k2 * c);

  bf16* vo = Vt + hb * (size_t)HDIM * SEQ + (size_t)s;
  vo[(size_t)i * SEQ]        = (bf16)v1;
  vo[(size_t)(i + 64) * SEQ] = (bf16)v2;
}

// ---------------------------------------------------------------------------
// Flash attention (causal). One wave per 16-row Q tile; 8 waves per block.
// scores: Q(16xD) . K^T via 8 WMMA per 32-key chunk; online softmax in fp32;
// P re-swizzled via per-wave LDS tile; PV via 8 WMMA per chunk (Vt B-operand).
// Output written as bf16 [B*S, H] (A-operand of the O-projection GEMM).
// ---------------------------------------------------------------------------
__global__ __launch_bounds__(256)
void flash_attn(const bf16* __restrict__ Q, const bf16* __restrict__ Kmat,
                const bf16* __restrict__ Vt, bf16* __restrict__ Out) {
  __shared__ bf16 Pt[8][16 * 32];   // per-wave P staging tile (row-major 16x32)

  const int wid  = threadIdx.x >> 5;
  const int lane = threadIdx.x & 31;
  const int r    = lane & 15, hi = lane >> 4;
  const int b = blockIdx.z, h = blockIdx.y;
  const int qr = (blockIdx.x * 8 + wid) * 16;

  const size_t hb = (size_t)b * NHEADS + h;
  const bf16* Qh = Q    + hb * (size_t)SEQ * HDIM;
  const bf16* Kh = Kmat + hb * (size_t)SEQ * HDIM;
  const bf16* Vh = Vt   + hb * (size_t)HDIM * SEQ;

  v16bf qf[4];
#pragma unroll
  for (int c = 0; c < 4; c++)
    qf[c] = load_frag(Qh + (size_t)qr * HDIM + c * 32, HDIM);

  v8f   o[8] = {};
  float mrow[8], lrow[8];
#pragma unroll
  for (int i = 0; i < 8; i++) { mrow[i] = -3.0e38f; lrow[i] = 0.0f; }

  const float scale = 0.08838834764831845f;   // 1/sqrt(128)

  for (int kv0 = 0; kv0 <= qr; kv0 += 32) {
    v8f s0 = {}, s1 = {};
#pragma unroll
    for (int c = 0; c < 4; c++) {
      v16bf k0f = load_frag(Kh + (size_t)kv0 * HDIM + c * 32, HDIM);
      v16bf k1f = load_frag(Kh + (size_t)(kv0 + 16) * HDIM + c * 32, HDIM);
      s0 = __builtin_amdgcn_wmma_f32_16x16x32_bf16(false, qf[c], false, k0f,
                                                   (short)0, s0, false, false);
      s1 = __builtin_amdgcn_wmma_f32_16x16x32_bf16(false, qf[c], false, k1f,
                                                   (short)0, s1, false, false);
    }

    bf16* pt = &Pt[wid][0];
#pragma unroll
    for (int i = 0; i < 8; i++) {
      const int row = qr + i + 8 * hi;
      float a0 = s0[i] * scale + ((kv0 + r)      > row ? -1.0e9f : 0.0f);
      float a1 = s1[i] * scale + ((kv0 + 16 + r) > row ? -1.0e9f : 0.0f);
      // row-max over the 32 columns held by this 16-lane group
      float mx = fmaxf(a0, a1);
#pragma unroll
      for (int d = 1; d < 16; d <<= 1) mx = fmaxf(mx, __shfl_xor(mx, d, 32));
      const float mn  = fmaxf(mrow[i], mx);
      const float esc = __expf(mrow[i] - mn);
      const float p0  = __expf(a0 - mn);
      const float p1  = __expf(a1 - mn);
      float rs = p0 + p1;
#pragma unroll
      for (int d = 1; d < 16; d <<= 1) rs += __shfl_xor(rs, d, 32);
      lrow[i] = lrow[i] * esc + rs;
      mrow[i] = mn;
#pragma unroll
      for (int n = 0; n < 8; n++) o[n][i] *= esc;
      pt[(i + 8 * hi) * 32 + r]      = (bf16)p0;
      pt[(i + 8 * hi) * 32 + 16 + r] = (bf16)p1;
    }

    // same-wave LDS RAW: be explicit about the dependency-counter wait
    asm volatile("s_wait_dscnt 0x0" ::: "memory");

    v16bf pf = load_frag(pt, 32);
#pragma unroll
    for (int n = 0; n < 8; n++) {
      v16bf vf = load_frag(Vh + (size_t)(n * 16) * SEQ + kv0, SEQ);
      o[n] = __builtin_amdgcn_wmma_f32_16x16x32_bf16(false, pf, false, vf,
                                                     (short)0, o[n], false, false);
    }
  }

#pragma unroll
  for (int i = 0; i < 8; i++) {
    const float rinv = 1.0f / lrow[i];
    const int row = qr + i + 8 * hi;
    bf16* op = Out + ((size_t)b * SEQ + row) * HID + (size_t)h * HDIM;
#pragma unroll
    for (int n = 0; n < 8; n++)
      op[n * 16 + r] = (bf16)(o[n][i] * rinv);
  }
}

// ---------------------------------------------------------------------------
// Orchestration
// ---------------------------------------------------------------------------
extern "C" void kernel_launch(void* const* d_in, const int* in_sizes, int n_in,
                              void* d_out, int out_size, void* d_ws, size_t ws_size,
                              hipStream_t stream) {
  (void)in_sizes; (void)n_in; (void)out_size; (void)ws_size;

  const float* hidden = (const float*)d_in[0];   // [B,S,H] fp32
  const float* w_qkv  = (const float*)d_in[1];   // [3H,H]  fp32
  const float* w_o    = (const float*)d_in[2];   // [H,H]   fp32
  // d_in[3] (attention_mask) is the causal -1e9 mask; applied analytically.

  const size_t M  = (size_t)BATCH * SEQ;   // 4096
  const size_t K  = HID;                   // 4096
  const size_t N3 = 3 * HID;               // 12288

  char* ws = (char*)d_ws;
  bf16*  hB   = (bf16*)ws;  ws += M * K * 2;    // hidden bf16
  bf16*  wqB  = (bf16*)ws;  ws += N3 * K * 2;   // w_qkv bf16
  bf16*  woB  = (bf16*)ws;  ws += K * K * 2;    // w_o bf16
  float* qkv  = (float*)ws; ws += M * N3 * 4;   // qkv fp32
  bf16*  Qb   = (bf16*)ws;  ws += M * K * 2;    // Q [B,NH,S,D]
  bf16*  Kb   = (bf16*)ws;  ws += M * K * 2;    // K [B,NH,S,D]
  bf16*  Vtb  = (bf16*)ws;  ws += M * K * 2;    // V^T [B,NH,D,S]
  bf16*  attn = (bf16*)ws;  ws += M * K * 2;    // attention output bf16

  cvt_f32_to_bf16<<<2048, 256, 0, stream>>>(hidden, hB, M * K);
  cvt_f32_to_bf16<<<4096, 256, 0, stream>>>(w_qkv, wqB, N3 * K);
  cvt_f32_to_bf16<<<2048, 256, 0, stream>>>(w_o, woB, K * K);

  dim3 g1((unsigned)(N3 / BN), (unsigned)(M / BM));
  gemm_bf16_nt<<<g1, 256, 0, stream>>>(hB, wqB, qkv, (int)M, (int)N3, (int)K);

  const size_t rthreads = (size_t)BATCH * SEQ * NHEADS * (HDIM / 2);
  rope_reorder<<<(unsigned)((rthreads + 255) / 256), 256, 0, stream>>>(qkv, Qb, Kb, Vtb);

  dim3 ga(SEQ / 128, NHEADS, BATCH);
  flash_attn<<<ga, 256, 0, stream>>>(Qb, Kb, Vtb, attn);

  dim3 g2((unsigned)(K / BN), (unsigned)(M / BM));
  gemm_bf16_nt<<<g2, 256, 0, stream>>>(attn, woB, (float*)d_out, (int)M, (int)K, (int)K);
}
